// MambaSequenceClassification_23261542875726
// MI455X (gfx1250) — matrix operand order repaired and checked by reading, not compiled, not run on hardware
//
#include <hip/hip_runtime.h>
#include <math.h>

// ---------------- model dims (from reference) ----------------
#define VOCAB     16
#define D_MODEL   512
#define NLAYERS   4
#define NCLASSES  2
#define D_STATE   128
#define D_CONV    4
#define HEADDIM   64
#define D_INNER   1024              // EXPAND * D_MODEL
#define NHEADS    16                // D_INNER / HEADDIM
#define CONV_DIM  1280              // D_INNER + 2*D_STATE
#define D_IN_PROJ 2320              // 2*D_INNER + 2*D_STATE + NHEADS
#define BATCH     8
#define SEQ       1024
#define EPS       1e-5f

typedef __attribute__((ext_vector_type(16))) __bf16 v16bf;
typedef __attribute__((ext_vector_type(4)))  __bf16 bf16x4;
typedef __attribute__((ext_vector_type(8)))  float  v8f;

// ---------------- embedding + positional ----------------
__global__ void embed_kernel(const int* __restrict__ tok, const float* __restrict__ emb,
                             const float* __restrict__ pos, float* __restrict__ x) {
  int idx = blockIdx.x * blockDim.x + threadIdx.x;      // over B*SEQ*D_MODEL
  if (idx >= BATCH * SEQ * D_MODEL) return;
  int d = idx % D_MODEL;
  int l = (idx / D_MODEL) % SEQ;
  int b = idx / (D_MODEL * SEQ);
  int t = tok[b * SEQ + l];
  x[idx] = emb[t * D_MODEL + d] + pos[l * D_MODEL + d];
}

// ---------------- WMMA bf16 GEMM: C[M,N] = A[M,K] * W[N,K]^T (+bias)(+res)(*mask)(gelu) ----
// block = 256 threads = 8 waves; block tile = 64(M) x 128(N); K stepped by 32.
// Wave w owns a 16-wide N slice; per K-step one B fragment is reused across 4
// M-subtiles -> 4x v_wmma_f32_16x16x32_bf16 per step.
// OOB rows are index-CLAMPED (not zero-filled): an OOB A row only pollutes an
// unstored C row; an OOB W row only pollutes an unstored C column. This keeps
// every global load unconditional (no exec-mask branches, batched waits).
// Global loads for step k+1 are issued before the WMMA phase of step k
// (register double-buffering) so VMEM latency overlaps matrix math.
#define TLM 64
#define TLN 128
#define LDST 40   // LDS row stride in bf16 (80 B = 20 dwords: 16B-aligned, bank-spread)

__global__ __launch_bounds__(256)
void wmma_gemm_bf16(const float* __restrict__ A, const float* __restrict__ W,
                    const float* __restrict__ bias, const float* __restrict__ res,
                    const float* __restrict__ mask, float* __restrict__ C,
                    int M, int N, int K, int act) {
  __shared__ __bf16 As[TLM][LDST];   // [m][k]
  __shared__ __bf16 Bs[TLN][LDST];   // [n][k]
  const int tid  = threadIdx.x;
  const int lane = tid & 31;
  const int w    = tid >> 5;
  const int half = lane >> 4;
  const int l16  = lane & 15;
  const int tileM = blockIdx.y * TLM;
  const int tileN = blockIdx.x * TLN;

  v8f acc[4] = {};

  // staging coordinates: float4 granularity, 8 float4 per 32-wide K row
  const int r0 = tid >> 3;           // 0..31
  const int c4 = tid & 7;            // float4 column within K-step

  // clamped row base pointers (row index never OOB; values may be garbage for
  // rows >= M/N but those only reach masked-out outputs)
  const float* aRow[2];
  const float* bRow[4];
#pragma unroll
  for (int j = 0; j < 2; ++j) {
    int m = tileM + r0 + j * 32; m = (m < M) ? m : (M - 1);
    aRow[j] = A + (size_t)m * K + c4 * 4;
  }
#pragma unroll
  for (int j = 0; j < 4; ++j) {
    int n = tileN + r0 + j * 32; n = (n < N) ? n : (N - 1);
    bRow[j] = W + (size_t)n * K + c4 * 4;
  }

  float4 aReg[2], bReg[4];
#pragma unroll
  for (int j = 0; j < 2; ++j) aReg[j] = *(const float4*)(aRow[j]);
#pragma unroll
  for (int j = 0; j < 4; ++j) bReg[j] = *(const float4*)(bRow[j]);

  for (int k0 = 0; k0 < K; k0 += 32) {
    // convert fp32 -> bf16, store 8B chunks to LDS
#pragma unroll
    for (int j = 0; j < 2; ++j) {
      bf16x4 p;
      p[0] = (__bf16)aReg[j].x; p[1] = (__bf16)aReg[j].y;
      p[2] = (__bf16)aReg[j].z; p[3] = (__bf16)aReg[j].w;
      *(bf16x4*)&As[r0 + j * 32][c4 * 4] = p;
    }
#pragma unroll
    for (int j = 0; j < 4; ++j) {
      bf16x4 p;
      p[0] = (__bf16)bReg[j].x; p[1] = (__bf16)bReg[j].y;
      p[2] = (__bf16)bReg[j].z; p[3] = (__bf16)bReg[j].w;
      *(bf16x4*)&Bs[r0 + j * 32][c4 * 4] = p;
    }
    __syncthreads();

    // prefetch next K-step into registers while WMMAs run
    if (k0 + 32 < K) {
#pragma unroll
      for (int j = 0; j < 2; ++j) aReg[j] = *(const float4*)(aRow[j] + k0 + 32);
#pragma unroll
      for (int j = 0; j < 4; ++j) bReg[j] = *(const float4*)(bRow[j] + k0 + 32);
    }

    // B fragment (32x16, K x N): N = lane&15, K = e + 8*half (e<8), e + 8 + 8*half (e>=8)
    v16bf bv;
#pragma unroll
    for (int e = 0; e < 16; ++e) bv[e] = Bs[w * 16 + l16][(e < 8) ? (e + half * 8) : (e + 8 + half * 8)];

#pragma unroll
    for (int ms = 0; ms < 4; ++ms) {
      // A fragment (16x32, M x K): M = lane&15, K split across lane halves
      v16bf av;
#pragma unroll
      for (int e = 0; e < 16; ++e) {
        int ka = (e < 8) ? (e + half * 8) : (e + 8 + half * 8);
        av[e] = As[ms * 16 + l16][ka];
      }
      acc[ms] = __builtin_amdgcn_wmma_f32_16x16x32_bf16(
          /*neg_a=*/false, av, /*neg_b=*/false, bv,
          /*c_mod=*/(short)0, acc[ms], /*reuse_a=*/false, /*reuse_b=*/false);
    }
    __syncthreads();
  }

  // epilogue: C layout — VGPR r holds M = r + 8*half, N = lane&15
  int n = tileN + w * 16 + l16;
  if (n < N) {
#pragma unroll
    for (int ms = 0; ms < 4; ++ms) {
#pragma unroll
      for (int r = 0; r < 8; ++r) {
        int m = tileM + ms * 16 + r + half * 8;
        if (m < M) {
          float v = acc[ms][r];
          if (bias) v += bias[n];
          if (res)  v += res[(size_t)m * N + n];
          if (act == 1) {  // tanh-approx gelu (jax.nn.gelu default)
            float x3 = v * v * v;
            v = 0.5f * v * (1.0f + tanhf(0.7978845608028654f * (v + 0.044715f * x3)));
          }
          if (mask) v *= mask[m];
          C[(size_t)m * N + n] = v;
        }
      }
    }
  }
}

// ---------------- layernorm (+mask) in place over D_MODEL=512, block=256 ----------------
__global__ __launch_bounds__(256)
void layernorm_kernel(float* __restrict__ x, const float* __restrict__ g,
                      const float* __restrict__ b, const float* __restrict__ mask) {
  __shared__ float red[8];
  size_t row = blockIdx.x;
  float* xr = x + row * D_MODEL;
  int tid = threadIdx.x;
  float v0 = xr[tid], v1 = xr[tid + 256];
  float s = v0 + v1;
  for (int o = 16; o > 0; o >>= 1) s += __shfl_xor(s, o, 32);
  if ((tid & 31) == 0) red[tid >> 5] = s;
  __syncthreads();
  float tot = 0.0f;
#pragma unroll
  for (int i = 0; i < 8; ++i) tot += red[i];
  float mu = tot / (float)D_MODEL;
  float d0 = v0 - mu, d1 = v1 - mu;
  float ss = d0 * d0 + d1 * d1;
  __syncthreads();
  for (int o = 16; o > 0; o >>= 1) ss += __shfl_xor(ss, o, 32);
  if ((tid & 31) == 0) red[tid >> 5] = ss;
  __syncthreads();
  float tv = 0.0f;
#pragma unroll
  for (int i = 0; i < 8; ++i) tv += red[i];
  float inv = rsqrtf(tv / (float)D_MODEL + EPS);
  float mk = mask[row];
  xr[tid]       = (d0 * inv * g[tid] + b[tid]) * mk;
  xr[tid + 256] = (d1 * inv * g[tid + 256] + b[tid + 256]) * mk;
}

// ---------------- depthwise causal conv (k=4) + silu ----------------
__global__ void conv_silu_kernel(const float* __restrict__ zx, const float* __restrict__ cw,
                                 const float* __restrict__ cb, float* __restrict__ xBC) {
  int idx = blockIdx.x * blockDim.x + threadIdx.x;   // over B*SEQ*CONV_DIM
  if (idx >= BATCH * SEQ * CONV_DIM) return;
  int c = idx % CONV_DIM;
  int l = (idx / CONV_DIM) % SEQ;
  int b = idx / (CONV_DIM * SEQ);
  float acc = cb[c];
#pragma unroll
  for (int k = 0; k < D_CONV; ++k) {
    int li = l - (D_CONV - 1) + k;
    if (li >= 0)
      acc += zx[((size_t)(b * SEQ + li)) * D_IN_PROJ + D_INNER + c] * cw[c * D_CONV + k];
  }
  float sig = 1.0f / (1.0f + expf(-acc));
  xBC[idx] = acc * sig;
}

// ---------------- SSM sequential scan: one block per (batch, head) ----------------
// 64x128 state in registers: thread (p = tid>>2) owns 32 of the 128 state cols.
__global__ __launch_bounds__(256)
void ssm_scan_kernel(const float* __restrict__ zx, const float* __restrict__ xBC,
                     const float* __restrict__ dt_bias, const float* __restrict__ A_log,
                     const float* __restrict__ Dp, float* __restrict__ y) {
  __shared__ float sx[HEADDIM];
  __shared__ float sB[D_STATE];
  __shared__ float sC[D_STATE];
  const int b  = blockIdx.x / NHEADS;
  const int hh = blockIdx.x % NHEADS;
  const int tid = threadIdx.x;
  const int p  = tid >> 2;
  const int nq = tid & 3;
  const float A   = -expf(A_log[hh]);
  const float dtb = dt_bias[hh];
  const float Dv  = Dp[hh];
  float h[32];
#pragma unroll
  for (int j = 0; j < 32; ++j) h[j] = 0.0f;

  for (int t = 0; t < SEQ; ++t) {
    size_t row = (size_t)b * SEQ + t;
    const float* xr = xBC + row * CONV_DIM;
    if (tid < HEADDIM)        sx[tid] = xr[hh * HEADDIM + tid];
    if (tid < D_STATE)        sB[tid] = xr[D_INNER + tid];
    else if (tid < 2*D_STATE) sC[tid - D_STATE] = xr[D_INNER + D_STATE + (tid - D_STATE)];
    float dts = zx[row * D_IN_PROJ + (2 * D_INNER + 2 * D_STATE) + hh] + dtb;
    float dt  = (dts > 20.0f) ? dts : log1pf(expf(dts));
    float dA  = expf(dt * A);
    __syncthreads();
    float xv   = sx[p];
    float coef = dt * xv;
    float acc  = 0.0f;
#pragma unroll
    for (int j = 0; j < 32; ++j) {
      int n = nq * 32 + j;
      h[j] = h[j] * dA + coef * sB[n];
      acc += h[j] * sC[n];
    }
    acc += __shfl_xor(acc, 1, 32);
    acc += __shfl_xor(acc, 2, 32);
    if (nq == 0) y[(row * NHEADS + hh) * HEADDIM + p] = acc + Dv * xv;
    __syncthreads();
  }
}

// ---------------- gate (y * silu(z)) + RMSNorm over D_INNER=1024 ----------------
__global__ __launch_bounds__(256)
void gate_rmsnorm_kernel(const float* __restrict__ y, const float* __restrict__ zx,
                         const float* __restrict__ nw, float* __restrict__ g) {
  __shared__ float red[8];
  size_t row = blockIdx.x;
  const float* yr = y + row * D_INNER;
  const float* zr = zx + row * D_IN_PROJ;   // z = first D_INNER of zxbcdt
  float vals[4];
  float ss = 0.0f;
#pragma unroll
  for (int i = 0; i < 4; ++i) {
    int e = threadIdx.x + i * 256;
    float z = zr[e];
    float v = yr[e] * (z / (1.0f + expf(-z)));
    vals[i] = v;
    ss += v * v;
  }
  for (int o = 16; o > 0; o >>= 1) ss += __shfl_xor(ss, o, 32);
  if ((threadIdx.x & 31) == 0) red[threadIdx.x >> 5] = ss;
  __syncthreads();
  float tot = 0.0f;
#pragma unroll
  for (int i = 0; i < 8; ++i) tot += red[i];
  float scale = rsqrtf(tot / (float)D_INNER + EPS);
#pragma unroll
  for (int i = 0; i < 4; ++i) {
    int e = threadIdx.x + i * 256;
    g[row * D_INNER + e] = vals[i] * scale * nw[e];
  }
}

// ---------------- (mean + max)/2 pooling over L ----------------
__global__ void pool_kernel(const float* __restrict__ x, float* __restrict__ pooled) {
  int idx = blockIdx.x * blockDim.x + threadIdx.x;   // over B*D_MODEL
  if (idx >= BATCH * D_MODEL) return;
  int d = idx % D_MODEL;
  int b = idx / D_MODEL;
  float s = 0.0f, mx = -1e30f;
  const float* base = x + ((size_t)b * SEQ) * D_MODEL + d;
  for (int l = 0; l < SEQ; ++l) {
    float v = base[(size_t)l * D_MODEL];
    s += v;
    mx = fmaxf(mx, v);
  }
  pooled[idx] = 0.5f * (s / (float)SEQ + mx);
}

// ---------------- final tiny classifier (8x2) ----------------
__global__ void cls2_kernel(const float* __restrict__ h, const float* __restrict__ w,
                            const float* __restrict__ bb, float* __restrict__ out) {
  int idx = threadIdx.x;
  if (idx >= BATCH * NCLASSES) return;
  int c = idx % NCLASSES, b = idx / NCLASSES;
  float acc = bb[c];
  for (int d = 0; d < D_MODEL / 2; ++d)
    acc += h[b * (D_MODEL / 2) + d] * w[c * (D_MODEL / 2) + d];
  out[idx] = acc;
}

// ---------------- driver ----------------
extern "C" void kernel_launch(void* const* d_in, const int* in_sizes, int n_in,
                              void* d_out, int out_size, void* d_ws, size_t ws_size,
                              hipStream_t stream) {
  const int*   tokens     = (const int*)d_in[0];
  const float* mask       = (const float*)d_in[1];
  const float* emb        = (const float*)d_in[2];
  const float* pos        = (const float*)d_in[3];
  const float* inp_w      = (const float*)d_in[4];
  const float* inp_b      = (const float*)d_in[5];
  const float* ln_g       = (const float*)d_in[6];
  const float* ln_b       = (const float*)d_in[7];
  const float* in_proj_w  = (const float*)d_in[8];
  const float* conv_w     = (const float*)d_in[9];
  const float* conv_b     = (const float*)d_in[10];
  const float* dt_bias    = (const float*)d_in[11];
  const float* A_log      = (const float*)d_in[12];
  const float* Dparam     = (const float*)d_in[13];
  const float* norm_w     = (const float*)d_in[14];
  const float* out_proj_w = (const float*)d_in[15];
  const float* pooler_w   = (const float*)d_in[16];
  const float* pooler_b   = (const float*)d_in[17];
  const float* cls_w1     = (const float*)d_in[18];
  const float* cls_b1     = (const float*)d_in[19];
  const float* cls_w2     = (const float*)d_in[20];
  const float* cls_b2     = (const float*)d_in[21];

  float* ws = (float*)d_ws;
  const size_t BL = (size_t)BATCH * SEQ;     // 8192 rows
  float* x0  = ws;                           // BL*512
  float* x   = x0  + BL * D_MODEL;           // BL*512
  float* zx  = x   + BL * D_MODEL;           // BL*2320
  float* xbc = zx  + BL * D_IN_PROJ;         // BL*1280
  float* yb  = xbc + BL * CONV_DIM;          // BL*1024
  float* gb  = yb  + BL * D_INNER;           // BL*1024
  float* pooled = gb + BL * D_INNER;         // 8*512
  float* h1  = pooled + BATCH * D_MODEL;     // 8*512
  float* h2  = h1 + BATCH * D_MODEL;         // 8*256

  auto gemm = [&](const float* A, const float* W, const float* bias, const float* res,
                  const float* msk, float* C, int M, int N, int K, int act) {
    dim3 grid((N + TLN - 1) / TLN, (M + TLM - 1) / TLM);
    wmma_gemm_bf16<<<grid, 256, 0, stream>>>(A, W, bias, res, msk, C, M, N, K, act);
  };

  {
    int n = BATCH * SEQ * D_MODEL;
    embed_kernel<<<(n + 255) / 256, 256, 0, stream>>>(tokens, emb, pos, x0);
  }
  gemm(x0, inp_w, inp_b, nullptr, nullptr, x, (int)BL, D_MODEL, D_MODEL, 0);
  layernorm_kernel<<<(int)BL, 256, 0, stream>>>(x, ln_g, ln_b, mask);

  for (int i = 0; i < NLAYERS; ++i) {
    gemm(x, in_proj_w + (size_t)i * D_IN_PROJ * D_MODEL,
         nullptr, nullptr, nullptr, zx, (int)BL, D_IN_PROJ, D_MODEL, 0);
    {
      int n = BATCH * SEQ * CONV_DIM;
      conv_silu_kernel<<<(n + 255) / 256, 256, 0, stream>>>(
          zx, conv_w + (size_t)i * CONV_DIM * D_CONV, conv_b + (size_t)i * CONV_DIM, xbc);
    }
    ssm_scan_kernel<<<BATCH * NHEADS, 256, 0, stream>>>(
        zx, xbc, dt_bias + i * NHEADS, A_log + i * NHEADS, Dparam + i * NHEADS, yb);
    gate_rmsnorm_kernel<<<(int)BL, 256, 0, stream>>>(yb, zx, norm_w + (size_t)i * D_INNER, gb);
    // out_proj + residual + mask, written in place over x (one thread per element)
    gemm(gb, out_proj_w + (size_t)i * D_MODEL * D_INNER,
         nullptr, x, mask, x, (int)BL, D_MODEL, D_INNER, 0);
  }

  pool_kernel<<<(BATCH * D_MODEL + 255) / 256, 256, 0, stream>>>(x, pooled);
  gemm(pooled, pooler_w, pooler_b, nullptr, nullptr, h1, BATCH, D_MODEL, D_MODEL, 1);
  gemm(h1, cls_w1, cls_b1, nullptr, nullptr, h2, BATCH, D_MODEL / 2, D_MODEL, 1);
  cls2_kernel<<<1, 32, 0, stream>>>(h2, cls_w2, cls_b2, (float*)d_out);
}
